// SampledQuantumInference_11845519802951
// MI455X (gfx1250) — matrix operand order repaired and checked by reading, not compiled, not run on hardware
//
#include <hip/hip_runtime.h>
#include <stdint.h>

typedef __attribute__((ext_vector_type(16))) _Float16 v16h;
typedef __attribute__((ext_vector_type(8)))  float    v8f;

#define NUM_UNIQUE  256
#define NUM_SAMPLES 4096
#define NUM_QUBITS  64
#define BATCH       1024
#define TILES       (NUM_SAMPLES / 16)   // 256 row tiles of 16 samples

union AF  { v16h h; uint32_t u[8]; };
union H16 { _Float16 f; uint16_t u; };

__global__ void __launch_bounds__(256) zero_acc_kernel(float* __restrict__ acc) {
    int i = blockIdx.x * blockDim.x + threadIdx.x;
    if (i < NUM_UNIQUE) acc[i] = 0.0f;
}

__global__ void __launch_bounds__(256) ising_energy_kernel(
    const int*   __restrict__ samples,   // [256][4096][64] bits {0,1}
    const float* __restrict__ bias,      // [64]
    const float* __restrict__ kern,      // [64][64]
    float*       __restrict__ acc)       // [256] energy sums
{
    // per-wave spin tile: 16 rows x 64 qubits of f16 (= 32 u32 per row)
    __shared__ uint32_t s_spin[8][16 * 32];

    const int tid  = threadIdx.x;
    const int wave = tid >> 5;
    const int lane = tid & 31;
    const int half = lane >> 4;     // 0/1: which half of the wave
    const int c16  = lane & 15;     // A row / B,D column within 16
    const int circ = blockIdx.x;

    // ---- B fragments: kernel matrix as f16, loaded once, resident in VGPRs ----
    // B 32x16 f16 layout (matches A): element e<8 -> K = 8*half + e,
    // e>=8 -> K = 16 + 8*half + (e-8), column N = lane%16. Pack pairs per VGPR.
    v16h Bf[8]; // [kb*4 + nb], kb = K-half (0..31 / 32..63), nb = 16-col block
    #pragma unroll
    for (int kb = 0; kb < 2; ++kb) {
        #pragma unroll
        for (int nb = 0; nb < 4; ++nb) {
            AF b;
            #pragma unroll
            for (int p = 0; p < 8; ++p) {
                const int k0 = kb * 32 + 8 * half + 2 * p + (p >= 4 ? 8 : 0);
                H16 h0; h0.f = (_Float16)kern[(k0 + 0) * 64 + nb * 16 + c16];
                H16 h1; h1.f = (_Float16)kern[(k0 + 1) * 64 + nb * 16 + c16];
                b.u[p] = (uint32_t)h0.u | ((uint32_t)h1.u << 16);
            }
            Bf[kb * 4 + nb] = b.h;
        }
    }

    float biasr[4];
    #pragma unroll
    for (int nb = 0; nb < 4; ++nb) biasr[nb] = bias[nb * 16 + c16];

    const size_t cbase = (size_t)circ * NUM_SAMPLES * NUM_QUBITS;
    float wave_sum = 0.0f;

    const int waveGlobal = blockIdx.y * 8 + wave;   // 64 waves per circuit
    const int waveStride = gridDim.y * 8;

    for (int t = waveGlobal; t < TILES; t += waveStride) {
        // ---- load 16x64 sample tile, each byte read exactly once (b128) ----
        // A 16x32 f16 layout: lane row = lane%16; groups of 8 K-values at
        // qbase = 8*half + 16*g  (g=0..3 across the two A fragments).
        const int row = c16;
        const int4* p4 = (const int4*)(samples + cbase + (size_t)(t * 16 + row) * NUM_QUBITS);
        AF a0, a1;
        #pragma unroll
        for (int g = 0; g < 4; ++g) {
            const int4 lo = p4[2 * half + 4 * g + 0];   // q = 8h+16g .. +3
            const int4 hi = p4[2 * half + 4 * g + 1];   // q = 8h+16g+4 .. +7
            // spin f16: +1 -> 0x3C00, -1 -> 0xBC00  (sign bit = sample bit)
            const uint32_t u0 = 0x3C003C00u | ((uint32_t)lo.x << 15) | ((uint32_t)lo.y << 31);
            const uint32_t u1 = 0x3C003C00u | ((uint32_t)lo.z << 15) | ((uint32_t)lo.w << 31);
            const uint32_t u2 = 0x3C003C00u | ((uint32_t)hi.x << 15) | ((uint32_t)hi.y << 31);
            const uint32_t u3 = 0x3C003C00u | ((uint32_t)hi.z << 15) | ((uint32_t)hi.w << 31);
            AF* dst = (g < 2) ? &a0 : &a1;
            const int o = (g & 1) * 4;
            dst->u[o + 0] = u0; dst->u[o + 1] = u1;
            dst->u[o + 2] = u2; dst->u[o + 3] = u3;
            // park spins in this wave's LDS slab (row-major 16x64 f16)
            uint4 st; st.x = u0; st.y = u1; st.z = u2; st.w = u3;
            ((uint4*)&s_spin[wave][0])[row * 8 + half + 2 * g] = st;
        }

        // ---- T = S x K : two chained WMMAs per 16-column block ----
        v8f d[4];
        #pragma unroll
        for (int nb = 0; nb < 4; ++nb) {
            v8f z = {0.f, 0.f, 0.f, 0.f, 0.f, 0.f, 0.f, 0.f};
            v8f t0 = __builtin_amdgcn_wmma_f32_16x16x32_f16(
                false, a0.h, false, Bf[0 * 4 + nb], (short)0, z, false, false);
            d[nb]  = __builtin_amdgcn_wmma_f32_16x16x32_f16(
                false, a1.h, false, Bf[1 * 4 + nb], (short)0, t0, false, false);
        }

        // ---- epilogue: sum_q s_q * (T_q + b_q), D rows = v + 8*half ----
        const _Float16* sp = (const _Float16*)&s_spin[wave][0];
        float lsum = 0.0f;
        #pragma unroll
        for (int nb = 0; nb < 4; ++nb) {
            const int   q  = nb * 16 + c16;
            const float bq = biasr[nb];
            #pragma unroll
            for (int v = 0; v < 8; ++v) {
                const int r = v + 8 * half;
                const float s = (float)sp[r * 64 + q];
                lsum += s * (d[nb][v] + bq);
            }
        }
        wave_sum += lsum;
    }

    // full-wave reduction (wave32), one atomic per wave
    #pragma unroll
    for (int m = 1; m < 32; m <<= 1)
        wave_sum += __shfl_xor(wave_sum, m, 32);
    if (lane == 0) atomicAdd(&acc[circ], wave_sum);
}

__global__ void __launch_bounds__(256) gather_mean_kernel(
    const float* __restrict__ acc,
    const int*   __restrict__ idx,
    float*       __restrict__ out)
{
    int i = blockIdx.x * blockDim.x + threadIdx.x;
    if (i < BATCH) out[i] = acc[idx[i]] * (1.0f / (float)NUM_SAMPLES);
}

extern "C" void kernel_launch(void* const* d_in, const int* in_sizes, int n_in,
                              void* d_out, int out_size, void* d_ws, size_t ws_size,
                              hipStream_t stream) {
    (void)in_sizes; (void)n_in; (void)out_size; (void)ws_size;
    const int*   samples = (const int*)  d_in[0];   // [256*4096*64] int32
    const int*   idx     = (const int*)  d_in[1];   // [1024] int32
    const float* bias    = (const float*)d_in[2];   // [64] f32
    const float* kern    = (const float*)d_in[3];   // [64*64] f32
    float* out = (float*)d_out;                     // [1024] f32
    float* acc = (float*)d_ws;                      // 256 f32 accumulators

    zero_acc_kernel<<<1, 256, 0, stream>>>(acc);

    dim3 grid(NUM_UNIQUE, 8);   // 256 circuits x 8 blocks; 8 waves/block -> 4 tiles/wave
    ising_energy_kernel<<<grid, 256, 0, stream>>>(samples, bias, kern, acc);

    gather_mean_kernel<<<(BATCH + 255) / 256, 256, 0, stream>>>(acc, idx, out);
}
